// LinearAttention_65738769432969
// MI455X (gfx1250) — matrix operand order repaired and checked by reading, not compiled, not run on hardware
//
#include <hip/hip_runtime.h>

// ---------------------------------------------------------------------------
// LinearAttention block for MI455X (gfx1250): bf16 WMMA GEMMs + async-LDS
// double-buffered staging.  B=16, DIM=512, H=W=64, HEADS=8, DH=64, INNER=512
// ---------------------------------------------------------------------------

#define BB     16
#define DIM    512
#define HH     64
#define WW     64
#define HW     4096              // H*W
#define NPOS   (BB * HW)         // 65536
#define HEADS  8
#define DH     64
#define BHD    (BB * HEADS)      // 128

typedef __attribute__((ext_vector_type(16))) __bf16 v16bf;
typedef __attribute__((ext_vector_type(8)))  float  v8f;

union ABpack { unsigned int u[8]; v16bf v; };

__device__ __forceinline__ unsigned short f2bf(float f) {
    unsigned int u = __float_as_uint(f);
    u = (u + 0x7FFFu + ((u >> 16) & 1u)) >> 16;   // round-to-nearest-even
    return (unsigned short)u;
}
__device__ __forceinline__ float bf2f(unsigned short h) {
    return __uint_as_float(((unsigned int)h) << 16);
}

// 16-bit A-matrix 16x32 per-lane K offsets (contiguous, matches kB), hi = lane>>4
__device__ __forceinline__ int kA(int j, int hi) {
    return (hi << 4) + (j << 1);
}
// 16-bit B-matrix 32x16 per-lane K offsets (contiguous, matches kA)
__device__ __forceinline__ int kB(int j, int hi) {
    return (hi << 4) + (j << 1);
}

// --- CDNA5 async global->LDS copy (ASYNCcnt-tracked) -----------------------
__device__ __forceinline__ void async_load_b64(unsigned lds_byte_addr,
                                               const void* gaddr) {
    asm volatile("global_load_async_to_lds_b64 %0, %1, off"
                 :: "v"(lds_byte_addr), "v"(gaddr) : "memory");
}
__device__ __forceinline__ void wait_asynccnt_le1() {
#if __has_builtin(__builtin_amdgcn_s_wait_asynccnt)
    __builtin_amdgcn_s_wait_asynccnt(1);
#else
    asm volatile("s_wait_asynccnt 0x1" ::: "memory");
#endif
}
__device__ __forceinline__ void wait_asynccnt_0() {
#if __has_builtin(__builtin_amdgcn_s_wait_asynccnt)
    __builtin_amdgcn_s_wait_asynccnt(0);
#else
    asm volatile("s_wait_asynccnt 0x0" ::: "memory");
#endif
}

// ---------------------------------------------------------------------------
// Channel LayerNorm of input fmap -> pos-major bf16 xt[(b*HW+pos)*512 + c]
// ---------------------------------------------------------------------------
__global__ void ln_in_kernel(const float* __restrict__ fmap,
                             const float* __restrict__ g,
                             unsigned short* __restrict__ xt) {
    size_t idx = (size_t)blockIdx.x * 256 + threadIdx.x;     // over B*HW
    int b = (int)(idx >> 12), pos = (int)(idx & 4095);
    const float* src = fmap + (((size_t)b * DIM) << 12) + pos;
    float s = 0.f, ss = 0.f;
    for (int c = 0; c < DIM; ++c) {
        float v = src[(size_t)c << 12];
        s += v; ss += v * v;
    }
    float mean = s * (1.f / DIM);
    float var  = ss * (1.f / DIM) - mean * mean;
    float inv  = rsqrtf(var + 1e-5f);
    unsigned short* dst = xt + idx * DIM;
    for (int c = 0; c < DIM; ++c)
        dst[c] = f2bf((src[(size_t)c << 12] - mean) * inv * g[c]);
}

// fp32 -> bf16 elementwise (weights)
__global__ void cvt_kernel(const float* __restrict__ src,
                           unsigned short* __restrict__ dst, int n) {
    int i = blockIdx.x * 256 + threadIdx.x;
    if (i < n) dst[i] = f2bf(src[i]);
}

// ---------------------------------------------------------------------------
// Pointwise 1x1 conv as WMMA GEMM: Out[o,posG] = sum_c W[o,c] * X[posG,c]
// Block tile: M=128 (8 waves x 16 o-rows) x N=32 positions.
// Activation tile (32 pos x 32 k bf16 = 2KB) is async-copied to LDS,
// double-buffered; all 8 waves consume the shared tile.  Weights stream
// directly from global (L2-resident).
// ---------------------------------------------------------------------------
__global__ void gemm_pw_kernel(const unsigned short* __restrict__ W,
                               const unsigned short* __restrict__ X,
                               unsigned short* __restrict__ Out) {
    __shared__ unsigned short bufB[2][32 * 32];            // 2 x 2KB, [pos][k]
    const int t = threadIdx.x;
    const int lane = t & 31, wave = t >> 5;                // 8 waves
    const int hi = lane >> 4, l16 = lane & 15;
    const int o0 = blockIdx.x * 128 + wave * 16;
    const int p0 = blockIdx.y * 32;                        // global position
    const unsigned short* arow = W + (size_t)(o0 + l16) * DIM;

    // staging assignment: 256 threads x 8B = 2KB tile
    const int sp = t >> 3;                                 // pos row 0..31
    const int sk = (t & 7) << 2;                           // k elem 0,4,..,28
    const unsigned short* gsrc0 = X + (size_t)(p0 + sp) * DIM + sk;

    async_load_b64((unsigned)(size_t)&bufB[0][sp * 32 + sk], gsrc0);

    v8f acc0 = {}, acc1 = {};
    for (int s = 0; s < 16; ++s) {                         // K = 512 / 32
        if (s + 1 < 16) {
            async_load_b64((unsigned)(size_t)&bufB[(s + 1) & 1][sp * 32 + sk],
                           gsrc0 + (s + 1) * 32);
            wait_asynccnt_le1();                           // stage s complete
        } else {
            wait_asynccnt_0();
        }
        __syncthreads();                                   // tile visible to all
        const unsigned short* bb = &bufB[s & 1][0];
        const int k0 = s * 32;
        ABpack a, b0, b1;
#pragma unroll
        for (int j = 0; j < 8; ++j)
            a.u[j] = *(const unsigned int*)(arow + k0 + kA(j, hi));
#pragma unroll
        for (int j = 0; j < 8; ++j) {
            b0.u[j] = *(const unsigned int*)(bb + l16 * 32 + kB(j, hi));
            b1.u[j] = *(const unsigned int*)(bb + (16 + l16) * 32 + kB(j, hi));
        }
        acc0 = __builtin_amdgcn_wmma_f32_16x16x32_bf16(
                   false, a.v, false, b0.v, (short)0, acc0, false, false);
        acc1 = __builtin_amdgcn_wmma_f32_16x16x32_bf16(
                   false, a.v, false, b1.v, (short)0, acc1, false, false);
        __syncthreads();                                   // done reading tile
    }
    const int bimg = p0 >> 12, pin = p0 & 4095;
#pragma unroll
    for (int r = 0; r < 8; ++r) {
        int o = o0 + (hi << 3) + r;
        size_t rowoff = ((size_t)(bimg * DIM + o)) << 12;
        Out[rowoff + pin + l16]      = f2bf(acc0[r]);
        Out[rowoff + pin + 16 + l16] = f2bf(acc1[r]);
    }
}

// ---------------------------------------------------------------------------
// Depthwise 3x3, SAME zero pad.  One thread per output element.
// ---------------------------------------------------------------------------
__global__ void dwconv_kernel(const unsigned short* __restrict__ in,
                              const float* __restrict__ w9,
                              unsigned short* __restrict__ out) {
    size_t idx = (size_t)blockIdx.x * 256 + threadIdx.x;  // over B*512*HW
    int p  = (int)(idx & 4095);
    int ch = (int)((idx >> 12) & 511);
    int x = p & 63, y = p >> 6;
    const float* wk = w9 + ch * 9;
    const unsigned short* base = in + (idx - (size_t)p);
    float acc = 0.f;
#pragma unroll
    for (int dy = -1; dy <= 1; ++dy) {
        int yy = y + dy; if ((unsigned)yy >= 64u) continue;
#pragma unroll
        for (int dx = -1; dx <= 1; ++dx) {
            int xx = x + dx; if ((unsigned)xx >= 64u) continue;
            acc += wk[(dy + 1) * 3 + (dx + 1)] * bf2f(base[yy * 64 + xx]);
        }
    }
    out[idx] = f2bf(acc);
}

// ---------------------------------------------------------------------------
// q softmax over feature dim d (64), * DH^-0.5; transpose to qt[(bh*HW+pos)*64+d]
// ---------------------------------------------------------------------------
__global__ void softmax_q_kernel(const unsigned short* __restrict__ dwq,
                                 unsigned short* __restrict__ qt) {
    size_t idx = (size_t)blockIdx.x * 256 + threadIdx.x;  // over BH*HW
    int pos = (int)(idx & 4095), bh = (int)(idx >> 12);
    const unsigned short* col = dwq + (((size_t)bh) << 18) + pos;
    float mx = -3.0e38f;
    for (int d = 0; d < DH; ++d) mx = fmaxf(mx, bf2f(col[(size_t)d << 12]));
    float s = 0.f;
    for (int d = 0; d < DH; ++d) s += __expf(bf2f(col[(size_t)d << 12]) - mx);
    float inv = 0.125f / s;                               // * DH^-0.5
    unsigned short* dst = qt + idx * DH;
    for (int d = 0; d < DH; ++d)
        dst[d] = f2bf(__expf(bf2f(col[(size_t)d << 12]) - mx) * inv);
}

// ---------------------------------------------------------------------------
// k softmax over sequence dim n (4096), in place; one 256-thread block per row.
// ---------------------------------------------------------------------------
__global__ void softmax_k_kernel(unsigned short* __restrict__ k) {
    __shared__ float row[4096];
    __shared__ float red[256];
    unsigned short* p = k + ((size_t)blockIdx.x << 12);   // row (bh*64+d)
    int t = threadIdx.x;
    float mx = -3.0e38f;
    for (int i = t; i < 4096; i += 256) { float v = bf2f(p[i]); row[i] = v; mx = fmaxf(mx, v); }
    red[t] = mx; __syncthreads();
    for (int s = 128; s > 0; s >>= 1) { if (t < s) red[t] = fmaxf(red[t], red[t + s]); __syncthreads(); }
    mx = red[0]; __syncthreads();
    float sum = 0.f;
    for (int i = t; i < 4096; i += 256) { float e = __expf(row[i] - mx); row[i] = e; sum += e; }
    red[t] = sum; __syncthreads();
    for (int s = 128; s > 0; s >>= 1) { if (t < s) red[t] += red[t + s]; __syncthreads(); }
    float inv = 1.f / red[0];
    for (int i = t; i < 4096; i += 256) p[i] = f2bf(row[i] * inv);
}

// ---------------------------------------------------------------------------
// ctx[bh,d,e] = sum_n ksm[bh,d,n] * vb[bh,e,n]   (K=4096 WMMA loop)
// One head per block, 16 waves each own a 16x16 tile of the 64x64 output.
// ---------------------------------------------------------------------------
__global__ void gemm_ctx_kernel(const unsigned short* __restrict__ K,
                                const unsigned short* __restrict__ V,
                                float* __restrict__ ctx) {
    int bh = blockIdx.x;
    int lane = threadIdx.x & 31, wave = threadIdx.x >> 5;  // 16 waves
    int hi = lane >> 4, l16 = lane & 15;
    int mt = wave & 3, nt = wave >> 2;
    const unsigned short* arow = K + (((size_t)(bh * DH + mt * 16 + l16)) << 12);
    const unsigned short* brow = V + (((size_t)(bh * DH + nt * 16 + l16)) << 12);
    v8f acc = {};
    for (int k0 = 0; k0 < HW; k0 += 32) {
        ABpack a, b;
#pragma unroll
        for (int j = 0; j < 8; ++j) {
            a.u[j] = *(const unsigned int*)(arow + k0 + kA(j, hi));
            b.u[j] = *(const unsigned int*)(brow + k0 + kB(j, hi));
        }
        acc = __builtin_amdgcn_wmma_f32_16x16x32_bf16(
                  false, a.v, false, b.v, (short)0, acc, false, false);
    }
#pragma unroll
    for (int r = 0; r < 8; ++r) {
        int d = mt * 16 + (hi << 3) + r, e = nt * 16 + l16;
        ctx[((size_t)bh * DH + d) * DH + e] = acc[r];
    }
}

// ---------------------------------------------------------------------------
// out = q @ ctx, fused SiLU, written pos-major: yt[(b*HW+pos)*512 + h*64+e]
// Block: 8 waves cover 128 positions; ctx staged transposed bf16 in LDS.
// ---------------------------------------------------------------------------
__global__ void gemm_out_kernel(const unsigned short* __restrict__ qt,
                                const float* __restrict__ ctx,
                                unsigned short* __restrict__ yt) {
    __shared__ unsigned short ctxT[DH * DH];               // [e][d], K-contiguous
    int bh = blockIdx.y, pbase = blockIdx.x * 128;
    int t = threadIdx.x;
    for (int i = t; i < DH * DH; i += 256) {
        int d = i >> 6, e = i & 63;
        ctxT[e * DH + d] = f2bf(ctx[((size_t)bh * DH + d) * DH + e]);
    }
    __syncthreads();
    int lane = t & 31, wave = t >> 5, hi = lane >> 4, l16 = lane & 15;
    const unsigned short* arow = qt + ((size_t)bh * HW + pbase + wave * 16 + l16) * DH;
    v8f acc[4] = {};
    for (int k0 = 0; k0 < DH; k0 += 32) {
        ABpack a;
#pragma unroll
        for (int j = 0; j < 8; ++j)
            a.u[j] = *(const unsigned int*)(arow + k0 + kA(j, hi));
#pragma unroll
        for (int nt = 0; nt < 4; ++nt) {
            ABpack b;
#pragma unroll
            for (int j = 0; j < 8; ++j)
                b.u[j] = *(const unsigned int*)(&ctxT[(nt * 16 + l16) * DH + k0 + kB(j, hi)]);
            acc[nt] = __builtin_amdgcn_wmma_f32_16x16x32_bf16(
                          false, a.v, false, b.v, (short)0, acc[nt], false, false);
        }
    }
    int b = bh >> 3, h = bh & 7;
#pragma unroll
    for (int nt = 0; nt < 4; ++nt) {
        int e = nt * 16 + l16;
#pragma unroll
        for (int r = 0; r < 8; ++r) {
            int pos = pbase + wave * 16 + (hi << 3) + r;
            float x = acc[nt][r];
            float s = x / (1.f + __expf(-x));              // SiLU
            yt[((size_t)(b * HW + pos)) * DIM + h * DH + e] = f2bf(s);
        }
    }
}

// ---------------------------------------------------------------------------
// Final channel LayerNorm: pre (b,o,pos) bf16 -> out (B,DIM,H,W) fp32
// ---------------------------------------------------------------------------
__global__ void ln_out_kernel(const unsigned short* __restrict__ pre,
                              const float* __restrict__ g,
                              float* __restrict__ out) {
    size_t idx = (size_t)blockIdx.x * 256 + threadIdx.x;   // over B*HW
    int b = (int)(idx >> 12), pos = (int)(idx & 4095);
    const unsigned short* src = pre + (((size_t)b * DIM) << 12) + pos;
    float s = 0.f, ss = 0.f;
    for (int c = 0; c < DIM; ++c) {
        float v = bf2f(src[(size_t)c << 12]);
        s += v; ss += v * v;
    }
    float mean = s * (1.f / DIM);
    float var  = ss * (1.f / DIM) - mean * mean;
    float inv  = rsqrtf(var + 1e-5f);
    for (int c = 0; c < DIM; ++c) {
        float v = (bf2f(src[(size_t)c << 12]) - mean) * inv * g[c];
        out[(((size_t)(b * DIM + c)) << 12) + pos] = v;
    }
}

// ---------------------------------------------------------------------------
extern "C" void kernel_launch(void* const* d_in, const int* in_sizes, int n_in,
                              void* d_out, int out_size, void* d_ws, size_t ws_size,
                              hipStream_t stream) {
    const float* fmap   = (const float*)d_in[0];
    const float* g_norm = (const float*)d_in[1];
    const float* q_w1   = (const float*)d_in[2];
    const float* q_dw   = (const float*)d_in[3];
    const float* k_w1   = (const float*)d_in[4];
    const float* k_dw   = (const float*)d_in[5];
    const float* v_w1   = (const float*)d_in[6];
    const float* v_dw   = (const float*)d_in[7];
    const float* out_w  = (const float*)d_in[8];
    const float* g_out  = (const float*)d_in[9];
    float* out = (float*)d_out;

    char* ws = (char*)d_ws;
    size_t off = 0;
    auto alloc = [&](size_t bytes) {
        size_t o = off; off += (bytes + 255) & ~(size_t)255; return o;
    };
    const size_t actB = (size_t)NPOS * DIM * 2;            // 64 MB bf16 tensors
    unsigned short* xt   = (unsigned short*)(ws + alloc(actB));   // LN out / later SiLU out
    unsigned short* tbuf = (unsigned short*)(ws + alloc(actB));   // GEMM out / final pre-LN
    unsigned short* dwq  = (unsigned short*)(ws + alloc(actB));   // depthwise q
    unsigned short* qt   = (unsigned short*)(ws + alloc(actB));   // softmax-q, pos-major
    unsigned short* ksm  = (unsigned short*)(ws + alloc(actB));   // depthwise k -> softmax-k
    unsigned short* vb   = (unsigned short*)(ws + alloc(actB));   // depthwise v (B operand)
    unsigned short* wq   = (unsigned short*)(ws + alloc((size_t)DIM * DIM * 2));
    unsigned short* wk   = (unsigned short*)(ws + alloc((size_t)DIM * DIM * 2));
    unsigned short* wv   = (unsigned short*)(ws + alloc((size_t)DIM * DIM * 2));
    unsigned short* wo   = (unsigned short*)(ws + alloc((size_t)DIM * DIM * 2));
    float* ctx           = (float*)(ws + alloc((size_t)BHD * DH * DH * 4));
    (void)ws_size; (void)in_sizes; (void)n_in; (void)out_size;

    const int WN = DIM * DIM;                              // 262144
    cvt_kernel<<<(WN + 255) / 256, 256, 0, stream>>>(q_w1, wq, WN);
    cvt_kernel<<<(WN + 255) / 256, 256, 0, stream>>>(k_w1, wk, WN);
    cvt_kernel<<<(WN + 255) / 256, 256, 0, stream>>>(v_w1, wv, WN);
    cvt_kernel<<<(WN + 255) / 256, 256, 0, stream>>>(out_w, wo, WN);

    ln_in_kernel<<<NPOS / 256, 256, 0, stream>>>(fmap, g_norm, xt);

    dim3 gGemm(DIM / 128, NPOS / 32);                      // (4, 2048)
    const int dwBlocks = (int)(((size_t)NPOS * DIM) / 256);

    // q path
    gemm_pw_kernel<<<gGemm, 256, 0, stream>>>(wq, xt, tbuf);
    dwconv_kernel<<<dwBlocks, 256, 0, stream>>>(tbuf, q_dw, dwq);
    softmax_q_kernel<<<(BHD * HW) / 256, 256, 0, stream>>>(dwq, qt);

    // k path
    gemm_pw_kernel<<<gGemm, 256, 0, stream>>>(wk, xt, tbuf);
    dwconv_kernel<<<dwBlocks, 256, 0, stream>>>(tbuf, k_dw, ksm);
    softmax_k_kernel<<<BHD * DH, 256, 0, stream>>>(ksm);

    // v path
    gemm_pw_kernel<<<gGemm, 256, 0, stream>>>(wv, xt, tbuf);
    dwconv_kernel<<<dwBlocks, 256, 0, stream>>>(tbuf, v_dw, vb);

    // attention
    gemm_ctx_kernel<<<BHD, 512, 0, stream>>>(ksm, vb, ctx);
    gemm_out_kernel<<<dim3(HW / 128, BHD), 256, 0, stream>>>(qt, ctx, xt); // xt reused as SiLU out

    // output projection + LN
    gemm_pw_kernel<<<gGemm, 256, 0, stream>>>(wo, xt, tbuf);
    ln_out_kernel<<<NPOS / 256, 256, 0, stream>>>(tbuf, g_out, out);
}
